// GRUBlock_26250840113809
// MI455X (gfx1250) — matrix operand order repaired and checked by reading, not compiled, not run on hardware
//
#include <hip/hip_runtime.h>

#define H    1024
#define S    4096
#define B    4
#define DI   1280
#define N1   (2*DI)     // 2560
#define EPS  1e-6f

typedef unsigned short ushort_t;
typedef __attribute__((ext_vector_type(16))) __bf16 v16bf;
typedef __attribute__((ext_vector_type(8)))  float  v8f;

union Frag { v16bf v; uint4 q[2]; };

__device__ __forceinline__ float bf2f(ushort_t u) {
    union { unsigned i; float f; } v; v.i = ((unsigned)u) << 16; return v.f;
}
__device__ __forceinline__ ushort_t f2bf(float f) {
    union { unsigned i; float f; } v; v.f = f;
    unsigned r = v.i + 0x7FFFu + ((v.i >> 16) & 1u);   // round-to-nearest-even
    return (ushort_t)(r >> 16);
}
__device__ __forceinline__ float sigf(float x) { return 1.0f / (1.0f + __expf(-x)); }
__device__ __forceinline__ float gfun(float x) { return (x >= 0.0f) ? (x + 0.5f) : sigf(x); }

// ---------------------------------------------------------------- RMSNorm
__global__ __launch_bounds__(256) void rmsnorm_kernel(const float* __restrict__ in,
                                                      const float* __restrict__ nw,
                                                      ushort_t* __restrict__ xout) {
    __shared__ float red[8];
    const int row = blockIdx.x;                 // 0..B*S-1
    const int t = threadIdx.x;
    const float* r = in + (size_t)row * H;
    float vals[4], ss = 0.0f;
#pragma unroll
    for (int i = 0; i < 4; ++i) { float v = r[t + i * 256]; vals[i] = v; ss += v * v; }
#pragma unroll
    for (int off = 16; off >= 1; off >>= 1) ss += __shfl_xor(ss, off, 32);
    if ((t & 31) == 0) red[t >> 5] = ss;
    __syncthreads();
    float tot = 0.0f;
#pragma unroll
    for (int i = 0; i < 8; ++i) tot += red[i];
    const float scale = rsqrtf(tot * (1.0f / H) + EPS);
    ushort_t* o = xout + (size_t)row * H;
#pragma unroll
    for (int i = 0; i < 4; ++i) {
        int h = t + i * 256;
        o[h] = f2bf(vals[i] * scale * nw[h]);
    }
}

// ---------------------------------------------------------------- pooled init + partial sums
__global__ void init_pool_kernel(float* __restrict__ pooled) {
    int t = threadIdx.x;
#pragma unroll
    for (int i = 0; i < 16; ++i) pooled[t + i * 256] = 0.0f;   // B*H = 4096
}

__global__ __launch_bounds__(256) void pool_kernel(const ushort_t* __restrict__ x,
                                                   float* __restrict__ pooled) {
    const int b = blockIdx.y;
    const int sc = blockIdx.x;                  // 32 chunks of 128 rows
    const int t = threadIdx.x;
    float acc[4] = {0.f, 0.f, 0.f, 0.f};
    const ushort_t* base = x + ((size_t)b * S + (size_t)sc * 128) * H + t * 4;
    for (int s = 0; s < 128; ++s) {
        uint2 d = *(const uint2*)(base + (size_t)s * H);
        union { uint2 d; ushort_t u[4]; } c; c.d = d;
        acc[0] += bf2f(c.u[0]); acc[1] += bf2f(c.u[1]);
        acc[2] += bf2f(c.u[2]); acc[3] += bf2f(c.u[3]);
    }
    const int h = t * 4;
#pragma unroll
    for (int i = 0; i < 4; ++i) atomicAdd(&pooled[b * H + h + i], acc[i]);
}

// ---------------------------------------------------------------- router (softmax probs)
__global__ __launch_bounds__(384) void router_kernel(const float* __restrict__ pooled,
                                                     const float* __restrict__ rw,
                                                     const float* __restrict__ rb,
                                                     float* __restrict__ probs) {
    __shared__ float lg[12];
    const int t = threadIdx.x, w = t >> 5, lane = t & 31;
    const int b = w / 3, e = w % 3;
    float s = 0.0f;
    for (int h = lane; h < H; h += 32) s += pooled[b * H + h] * rw[h * 3 + e];
#pragma unroll
    for (int off = 16; off >= 1; off >>= 1) s += __shfl_xor(s, off, 32);
    if (lane == 0) lg[w] = s * (1.0f / S) + rb[e];
    __syncthreads();
    if (t < B) {
        float a = lg[t * 3], bb = lg[t * 3 + 1], c = lg[t * 3 + 2];
        float mx = fmaxf(a, fmaxf(bb, c));
        float ea = __expf(a - mx), eb = __expf(bb - mx), ec = __expf(c - mx);
        float inv = 1.0f / (ea + eb + ec);
        probs[t * 3 + 0] = ea * inv; probs[t * 3 + 1] = eb * inv; probs[t * 3 + 2] = ec * inv;
    }
}

// ---------------------------------------------------------------- weight mixing (transposed, bf16)
// WiT[b][n][k] = sum_e p[b][e] * w_in[e][k][n]   (n in [0,2560), k in [0,1024))
__global__ __launch_bounds__(256) void combine_win_kernel(const float* __restrict__ w_in,
                                                          const float* __restrict__ probs,
                                                          ushort_t* __restrict__ WiT) {
    __shared__ float tile[32][33];
    const int b = blockIdx.z;
    const int n0 = blockIdx.x * 32;             // 80 tiles
    const int k0 = blockIdx.y * 32;             // 32 tiles
    const float p0 = probs[b * 3], p1 = probs[b * 3 + 1], p2 = probs[b * 3 + 2];
    const int t = threadIdx.x;
    const size_t ES = (size_t)H * N1;
#pragma unroll
    for (int r = 0; r < 4; ++r) {
        int id = t + 256 * r;
        int kk = id >> 5, nn = id & 31;
        size_t off = (size_t)(k0 + kk) * N1 + (n0 + nn);
        tile[kk][nn] = p0 * w_in[off] + p1 * w_in[off + ES] + p2 * w_in[off + 2 * ES];
    }
    __syncthreads();
#pragma unroll
    for (int r = 0; r < 4; ++r) {
        int id = t + 256 * r;
        int nn = id >> 5, kk = id & 31;
        WiT[((size_t)b * N1 + n0 + nn) * H + k0 + kk] = f2bf(tile[kk][nn]);
    }
}

// WoT[b][h][d] = sum_e p[b][e] * w_out[e][d][h]
__global__ __launch_bounds__(256) void combine_wo_kernel(const float* __restrict__ w_out,
                                                         const float* __restrict__ probs,
                                                         ushort_t* __restrict__ WoT) {
    __shared__ float tile[32][33];
    const int b = blockIdx.z;
    const int h0 = blockIdx.x * 32;             // 32 tiles
    const int d0 = blockIdx.y * 32;             // 40 tiles
    const float p0 = probs[b * 3], p1 = probs[b * 3 + 1], p2 = probs[b * 3 + 2];
    const int t = threadIdx.x;
    const size_t ES = (size_t)DI * H;
#pragma unroll
    for (int r = 0; r < 4; ++r) {
        int id = t + 256 * r;
        int dd = id >> 5, hh = id & 31;
        size_t off = (size_t)(d0 + dd) * H + (h0 + hh);
        tile[dd][hh] = p0 * w_out[off] + p1 * w_out[off + ES] + p2 * w_out[off + 2 * ES];
    }
    __syncthreads();
#pragma unroll
    for (int r = 0; r < 4; ++r) {
        int id = t + 256 * r;
        int hh = id >> 5, dd = id & 31;
        WoT[((size_t)b * H + h0 + hh) * DI + d0 + dd] = f2bf(tile[dd][hh]);
    }
}

__global__ void bias_kernel(const float* __restrict__ b_in, const float* __restrict__ b_out,
                            const float* __restrict__ probs,
                            float* __restrict__ bi, float* __restrict__ bo) {
    const int idx = blockIdx.x * 256 + threadIdx.x;
    const int NI = B * N1;                      // 10240
    if (idx < NI) {
        int b = idx / N1, n = idx % N1;
        bi[idx] = probs[b * 3] * b_in[n] + probs[b * 3 + 1] * b_in[N1 + n] +
                  probs[b * 3 + 2] * b_in[2 * N1 + n];
    } else if (idx < NI + B * H) {
        int j = idx - NI; int b = j / H, n = j % H;
        bo[j] = probs[b * 3] * b_out[n] + probs[b * 3 + 1] * b_out[H + n] +
                probs[b * 3 + 2] * b_out[2 * H + n];
    }
}

// ---------------------------------------------------------------- WMMA GEMM core
#define BM 128
#define BN 128
#define BK 32
#define LDT 40   // LDS row stride (elements): 80B rows -> 16B aligned, conflict-friendly

__device__ __forceinline__ void load_tile(const ushort_t* __restrict__ g, int ld,
                                          int row0, int k0, ushort_t* lds, int tid) {
#pragma unroll
    for (int r = 0; r < 2; ++r) {
        int c = tid + 256 * r;                  // 512 chunks of 8 bf16
        int row = c >> 2;
        int kc = (c & 3) * 8;
        const uint4* src = (const uint4*)(g + (size_t)(row0 + row) * ld + k0 + kc);
        *(uint4*)(lds + row * LDT + kc) = *src;
    }
}

__device__ __forceinline__ void mma_loop(const ushort_t* __restrict__ gA, int ldA,
                                         const ushort_t* __restrict__ gB, int ldB,
                                         int K, int blockM, int blockN,
                                         ushort_t* ldsA, ushort_t* ldsB, v8f acc[2][4]) {
    const int tid = threadIdx.x;
    const int wid = tid >> 5;
    const int lane = tid & 31;
    const int waveM = (wid >> 1) * 32;
    const int waveN = (wid & 1) * 64;
    const int lr = lane & 15;
    const int lh = lane >> 4;
    for (int k0 = 0; k0 < K; k0 += BK) {
        load_tile(gA, ldA, blockM, k0, ldsA, tid);
        load_tile(gB, ldB, blockN, k0, ldsB, tid);
        __syncthreads();
        Frag a[2], bf[4];
#pragma unroll
        for (int i = 0; i < 2; ++i) {
            // A 16x32 bf16 frag: lanes<16 hold K {0..7,16..23}, lanes>=16 hold K {8..15,24..31}
            const ushort_t* p = ldsA + (waveM + i * 16 + lr) * LDT + lh * 8;
            a[i].q[0] = *(const uint4*)(p);
            a[i].q[1] = *(const uint4*)(p + 16);
        }
#pragma unroll
        for (int j = 0; j < 4; ++j) {
            // B 32x16 bf16 frag: lanes<16 hold K {0..7,16..23}, lanes>=16 hold K {8..15,24..31}
            const ushort_t* p = ldsB + (waveN + j * 16 + lr) * LDT + lh * 8;
            bf[j].q[0] = *(const uint4*)(p);
            bf[j].q[1] = *(const uint4*)(p + 16);
        }
#pragma unroll
        for (int i = 0; i < 2; ++i)
#pragma unroll
            for (int j = 0; j < 4; ++j)
                acc[i][j] = __builtin_amdgcn_wmma_f32_16x16x32_bf16(
                    false, a[i].v, false, bf[j].v, (short)0, acc[i][j], false, false);
        __syncthreads();
    }
}

// GEMM1: zh = x(S,H) @ WiT^T(H,2DI) + bi  -> activations -> z/ht (bf16)
__global__ __launch_bounds__(256) void gemm1_kernel(const ushort_t* __restrict__ x,
                                                    const ushort_t* __restrict__ WiT,
                                                    const float* __restrict__ bi,
                                                    ushort_t* __restrict__ zbuf,
                                                    ushort_t* __restrict__ htbuf) {
    __shared__ ushort_t ldsA[BM * LDT];
    __shared__ ushort_t ldsB[BN * LDT];
    const int b = blockIdx.z;
    const int blockM = blockIdx.y * BM;
    const int blockN = blockIdx.x * BN;
    v8f acc[2][4] = {};
    mma_loop(x + (size_t)b * S * H, H, WiT + (size_t)b * N1 * H, H, H,
             blockM, blockN, ldsA, ldsB, acc);
    const int tid = threadIdx.x, wid = tid >> 5, lane = tid & 31;
    const int waveM = (wid >> 1) * 32, waveN = (wid & 1) * 64;
#pragma unroll
    for (int i = 0; i < 2; ++i)
#pragma unroll
        for (int j = 0; j < 4; ++j)
#pragma unroll
            for (int v = 0; v < 8; ++v) {
                int m = blockM + waveM + i * 16 + (lane >> 4) * 8 + v;
                int n = blockN + waveN + j * 16 + (lane & 15);
                float val = acc[i][j][v] + bi[b * N1 + n];
                if (n < DI) htbuf[((size_t)b * S + m) * DI + n] = f2bf(gfun(val));
                else        zbuf [((size_t)b * S + m) * DI + (n - DI)] = f2bf(sigf(val));
            }
}

// GRU scan: h_s = (1-z_s) h_{s-1} + z_s ht_s ; one thread per (b,d) channel
__global__ __launch_bounds__(256) void scan_kernel(const ushort_t* __restrict__ zbuf,
                                                   const ushort_t* __restrict__ htbuf,
                                                   ushort_t* __restrict__ hs,
                                                   float* __restrict__ out, size_t outOff) {
    const int c = blockIdx.x * 256 + threadIdx.x;   // 0..B*DI-1
    const int b = c / DI, d = c % DI;
    const size_t base = ((size_t)b * S) * DI + d;
    float h = 0.0f;
#pragma unroll 4
    for (int s = 0; s < S; ++s) {
        size_t o = base + (size_t)s * DI;
        float z  = bf2f(zbuf[o]);
        float ht = bf2f(htbuf[o]);
        h = (1.0f - z) * h + z * ht;
        hs[o] = f2bf(h);
    }
    out[outOff + c] = h;                            // new_state (fp32)
    if (c == 0) out[outOff + B * DI] = 0.0f;        // aux_loss
}

// GEMM2: out = hs(S,DI) @ WoT^T(DI,H) + bo + inputs (residual), fp32
__global__ __launch_bounds__(256) void gemm2_kernel(const ushort_t* __restrict__ hs,
                                                    const ushort_t* __restrict__ WoT,
                                                    const float* __restrict__ bo,
                                                    const float* __restrict__ inputs,
                                                    float* __restrict__ out) {
    __shared__ ushort_t ldsA[BM * LDT];
    __shared__ ushort_t ldsB[BN * LDT];
    const int b = blockIdx.z;
    const int blockM = blockIdx.y * BM;
    const int blockN = blockIdx.x * BN;
    v8f acc[2][4] = {};
    mma_loop(hs + (size_t)b * S * DI, DI, WoT + (size_t)b * H * DI, DI, DI,
             blockM, blockN, ldsA, ldsB, acc);
    const int tid = threadIdx.x, wid = tid >> 5, lane = tid & 31;
    const int waveM = (wid >> 1) * 32, waveN = (wid & 1) * 64;
#pragma unroll
    for (int i = 0; i < 2; ++i)
#pragma unroll
        for (int j = 0; j < 4; ++j)
#pragma unroll
            for (int v = 0; v < 8; ++v) {
                int m = blockM + waveM + i * 16 + (lane >> 4) * 8 + v;
                int n = blockN + waveN + j * 16 + (lane & 15);
                size_t o = ((size_t)b * S + m) * H + n;
                out[o] = acc[i][j][v] + bo[b * H + n] + inputs[o];
            }
}

// ---------------------------------------------------------------- launch
extern "C" void kernel_launch(void* const* d_in, const int* in_sizes, int n_in,
                              void* d_out, int out_size, void* d_ws, size_t ws_size,
                              hipStream_t stream) {
    const float* inputs   = (const float*)d_in[0];
    const float* norm_w   = (const float*)d_in[1];
    const float* router_w = (const float*)d_in[2];
    const float* router_b = (const float*)d_in[3];
    const float* w_in     = (const float*)d_in[4];
    const float* b_in     = (const float*)d_in[5];
    const float* w_out    = (const float*)d_in[6];
    const float* b_out    = (const float*)d_in[7];
    float* out = (float*)d_out;

    char* ws = (char*)d_ws;
    size_t off = 0;
    auto alloc = [&](size_t bytes) -> void* {
        void* p = ws + off;
        off += (bytes + 255) & ~(size_t)255;
        return p;
    };
    ushort_t* xbuf   = (ushort_t*)alloc((size_t)B * S * H * 2);    // 32 MB
    ushort_t* zbuf   = (ushort_t*)alloc((size_t)B * S * DI * 2);   // 40 MB
    ushort_t* htbuf  = (ushort_t*)alloc((size_t)B * S * DI * 2);   // 40 MB
    ushort_t* hsbuf  = (ushort_t*)alloc((size_t)B * S * DI * 2);   // 40 MB
    ushort_t* WiT    = (ushort_t*)alloc((size_t)B * N1 * H * 2);   // 20 MB
    ushort_t* WoT    = (ushort_t*)alloc((size_t)B * H * DI * 2);   // 10 MB
    float*    bi     = (float*)alloc((size_t)B * N1 * 4);
    float*    bo     = (float*)alloc((size_t)B * H * 4);
    float*    pooled = (float*)alloc((size_t)B * H * 4);
    float*    probs  = (float*)alloc((size_t)B * 3 * 4);

    init_pool_kernel<<<1, 256, 0, stream>>>(pooled);
    rmsnorm_kernel<<<B * S, 256, 0, stream>>>(inputs, norm_w, xbuf);
    pool_kernel<<<dim3(32, B), 256, 0, stream>>>(xbuf, pooled);
    router_kernel<<<1, 384, 0, stream>>>(pooled, router_w, router_b, probs);
    combine_win_kernel<<<dim3(N1 / 32, H / 32, B), 256, 0, stream>>>(w_in, probs, WiT);
    combine_wo_kernel<<<dim3(H / 32, DI / 32, B), 256, 0, stream>>>(w_out, probs, WoT);
    bias_kernel<<<(B * (N1 + H) + 255) / 256, 256, 0, stream>>>(b_in, b_out, probs, bi, bo);
    gemm1_kernel<<<dim3(N1 / BN, S / BM, B), 256, 0, stream>>>(xbuf, WiT, bi, zbuf, htbuf);
    scan_kernel<<<(B * DI) / 256, 256, 0, stream>>>(zbuf, htbuf, hsbuf, out, (size_t)B * S * H);
    gemm2_kernel<<<dim3(H / BN, S / BM, B), 256, 0, stream>>>(hsbuf, WoT, bo, inputs, out);
}
